// GemmaDecoderLayer_28827820491283
// MI455X (gfx1250) — matrix operand order repaired and checked by reading, not compiled, not run on hardware
//
#include <hip/hip_runtime.h>
#include <hip/hip_bf16.h>

// ---------------------------------------------------------------------------
// Gemma decoder layer for MI455X (gfx1250), wave32, WMMA bf16 (f32 accum).
//
// Roofline: ~1.1 TFLOP of GEMMs vs 437 MB of read-once fp32 weights ->
// memory-bound at ~19 us (23.3 TB/s). Weights are streamed fp32 ONCE from HBM,
// converted bf16 in-register (v_cvt_pk_bf16_f32), fed to v_wmma_f32_16x16x32_bf16.
// Grid is ordered M-fastest so the weight stripe (<=16 MB) + activations (32 MB)
// stay resident in the 192 MB L2 across all M-blocks -> weight traffic ~ W once.
// Each wave register-blocks 2x2 tiles (32x32) so every A/B fragment feeds 2 WMMAs.
// ---------------------------------------------------------------------------

#define NB  64        // batch
#define NS  64        // seq
#define ND  2048      // hidden dim
#define NH  8         // heads
#define NHD 256       // head dim
#define NI  16384     // intermediate
#define NM  (NB * NS) // 4096 token rows

typedef __bf16 bf16;
typedef __attribute__((ext_vector_type(16))) __bf16 bf16x16;
typedef __attribute__((ext_vector_type(8)))  __bf16 bf16x8;
typedef __attribute__((ext_vector_type(8)))  float  f32x8;

#define WMMA_BF16(a, b, c) \
  __builtin_amdgcn_wmma_f32_16x16x32_bf16(false, (a), false, (b), (short)0, (c), false, false)

// ---------------- fragment loaders (CDNA5 ISA 7.12.2 layouts) ----------------

// A 16x32 bf16 frag from row-major f32 matrix (lda in elements).
// lane 0-15: M=lane, K = kk+0..15; lane 16-31: M=lane-16, K = kk+16..31
__device__ __forceinline__ bf16x16 load_a(const float* A, int lda, int m0, int kk, int lane) {
  int row = m0 + (lane & 15);
  const float* p = A + (size_t)row * lda + kk + ((lane >> 4) << 4);
  const float4* p4 = (const float4*)p;
  float4 x0 = p4[0], x1 = p4[1];   // K offsets +0..+7
  float4 y0 = p4[2], y1 = p4[3];   // K offsets +8..+15
  bf16x16 a;
  a[0]  = (bf16)x0.x; a[1]  = (bf16)x0.y; a[2]  = (bf16)x0.z; a[3]  = (bf16)x0.w;
  a[4]  = (bf16)x1.x; a[5]  = (bf16)x1.y; a[6]  = (bf16)x1.z; a[7]  = (bf16)x1.w;
  a[8]  = (bf16)y0.x; a[9]  = (bf16)y0.y; a[10] = (bf16)y0.z; a[11] = (bf16)y0.w;
  a[12] = (bf16)y1.x; a[13] = (bf16)y1.y; a[14] = (bf16)y1.z; a[15] = (bf16)y1.w;
  return a;
}

// A frag from row-major bf16 matrix.
__device__ __forceinline__ bf16x16 load_a(const bf16* A, int lda, int m0, int kk, int lane) {
  int row = m0 + (lane & 15);
  const bf16* p = A + (size_t)row * lda + kk + ((lane >> 4) << 4);
  bf16x8 lo = *(const bf16x8*)p;         // K +0..+7
  bf16x8 hi = *(const bf16x8*)(p + 8);   // K +8..+15
  bf16x16 a;
#pragma unroll
  for (int j = 0; j < 8; ++j) { a[j] = lo[j]; a[8 + j] = hi[j]; }
  return a;
}

// B 32x16 frag where B[k][n] = W[n][k] (weight is [N][K] row-major, i.e. A@W^T).
// lane 0-15: N=lane, K = kk+0..15; lane 16-31: N=lane-16, K = kk+16..31
__device__ __forceinline__ bf16x16 load_b_wt(const float* W, int ldw, int n0, int kk, int lane) {
  int col = n0 + (lane & 15);
  const float* p = W + (size_t)col * ldw + kk + ((lane >> 4) << 4);
  const float4* p4 = (const float4*)p;
  float4 x0 = p4[0], x1 = p4[1], x2 = p4[2], x3 = p4[3];
  bf16x16 b;
  b[0]  = (bf16)x0.x; b[1]  = (bf16)x0.y; b[2]  = (bf16)x0.z; b[3]  = (bf16)x0.w;
  b[4]  = (bf16)x1.x; b[5]  = (bf16)x1.y; b[6]  = (bf16)x1.z; b[7]  = (bf16)x1.w;
  b[8]  = (bf16)x2.x; b[9]  = (bf16)x2.y; b[10] = (bf16)x2.z; b[11] = (bf16)x2.w;
  b[12] = (bf16)x3.x; b[13] = (bf16)x3.y; b[14] = (bf16)x3.z; b[15] = (bf16)x3.w;
  return b;
}

// B 32x16 frag from [K][N] row-major f32 matrix (used for P@V).
__device__ __forceinline__ bf16x16 load_b_kn(const float* V, int ldn, int n0, int kk, int lane) {
  int col = n0 + (lane & 15);
  int kb  = kk + ((lane >> 4) << 4);
  bf16x16 b;
#pragma unroll
  for (int j = 0; j < 16; ++j) b[j] = (bf16)V[(size_t)(kb + j) * ldn + col];
  return b;
}

// D 16x16 f32 store: VGPR r -> M = m0 + r + (lane<16 ? 0 : 8), N = n0 + lane%16
__device__ __forceinline__ void store_d(float* C, int ldc, int m0, int n0, int lane,
                                        f32x8 c, const float* resid) {
  int col   = n0 + (lane & 15);
  int rbase = m0 + ((lane >> 4) << 3);
#pragma unroll
  for (int r = 0; r < 8; ++r) {
    size_t idx = (size_t)(rbase + r) * ldc + col;
    C[idx] = resid ? (c[r] + resid[idx]) : c[r];
  }
}

__device__ __forceinline__ float gelu_tanh(float x) {
  return 0.5f * x * (1.0f + tanhf(0.7978845608028654f * (x + 0.044715f * x * x * x)));
}

// ---------------- kernels ----------------

__global__ __launch_bounds__(256)
void rmsnorm_kernel(const float* __restrict__ X, const float* __restrict__ w,
                    float* __restrict__ Y, int ncols) {
  __shared__ float red[8];
  int row = blockIdx.x;
  const float* x = X + (size_t)row * ncols;
  float ss = 0.f;
  for (int i = threadIdx.x; i < ncols; i += 256) { float v = x[i]; ss += v * v; }
#pragma unroll
  for (int off = 16; off > 0; off >>= 1) ss += __shfl_down(ss, off, 32);
  if ((threadIdx.x & 31) == 0) red[threadIdx.x >> 5] = ss;
  __syncthreads();
  if (threadIdx.x == 0) {
    float t = 0.f;
#pragma unroll
    for (int i = 0; i < 8; ++i) t += red[i];
    red[0] = rsqrtf(t / (float)ncols + 1e-6f);
  }
  __syncthreads();
  float sc = red[0];
  for (int i = threadIdx.x; i < ncols; i += 256)
    Y[(size_t)row * ncols + i] = x[i] * sc * (1.0f + w[i]);
}

// C[M,N] = A[M,K] @ W[N,K]^T (+resid).
// 256 threads = 8 waves. Block tile 32x256; wave tile 32x32 (2x2 WMMA tiles).
// blockIdx.x = M-tile (fastest) so concurrent blocks share one W stripe in L2.
template <typename AT>
__global__ __launch_bounds__(256)
void wmma_gemm(const AT* __restrict__ A, const float* __restrict__ W,
               const float* __restrict__ resid, float* __restrict__ C,
               int M, int N, int K) {
  int lane = threadIdx.x & 31, wave = threadIdx.x >> 5;
  int m0 = blockIdx.x * 32;
  int n0 = blockIdx.y * 256 + wave * 32;
  f32x8 c00 = {}, c01 = {}, c10 = {}, c11 = {};
  for (int kk = 0; kk < K; kk += 32) {
    bf16x16 a0 = load_a(A, K, m0,      kk, lane);   // shared by all 8 waves -> WGP$
    bf16x16 a1 = load_a(A, K, m0 + 16, kk, lane);
    bf16x16 b0 = load_b_wt(W, K, n0,      kk, lane);
    bf16x16 b1 = load_b_wt(W, K, n0 + 16, kk, lane);
    c00 = WMMA_BF16(a0, b0, c00);
    c01 = WMMA_BF16(a0, b1, c01);
    c10 = WMMA_BF16(a1, b0, c10);
    c11 = WMMA_BF16(a1, b1, c11);
  }
  store_d(C, N, m0,      n0,      lane, c00, resid);
  store_d(C, N, m0,      n0 + 16, lane, c01, resid);
  store_d(C, N, m0 + 16, n0,      lane, c10, resid);
  store_d(C, N, m0 + 16, n0 + 16, lane, c11, resid);
}

// act[M,N] = bf16( gelu(A@Wg^T) * (A@Wu^T) ).
// Block tile 32x128; wave: 2 M-tiles x 1 N-tile x 2 weight streams = 4 WMMAs
// per k-step with both A frags and both B frags reused twice.
__global__ __launch_bounds__(256)
void wmma_gateup(const float* __restrict__ A, const float* __restrict__ Wg,
                 const float* __restrict__ Wu, bf16* __restrict__ act,
                 int M, int N, int K) {
  int lane = threadIdx.x & 31, wave = threadIdx.x >> 5;
  int m0 = blockIdx.x * 32;
  int n0 = blockIdx.y * 128 + wave * 16;
  f32x8 g0 = {}, g1 = {}, u0 = {}, u1 = {};
  for (int kk = 0; kk < K; kk += 32) {
    bf16x16 a0 = load_a(A, K, m0,      kk, lane);
    bf16x16 a1 = load_a(A, K, m0 + 16, kk, lane);
    bf16x16 bg = load_b_wt(Wg, K, n0, kk, lane);
    bf16x16 bu = load_b_wt(Wu, K, n0, kk, lane);
    g0 = WMMA_BF16(a0, bg, g0);
    g1 = WMMA_BF16(a1, bg, g1);
    u0 = WMMA_BF16(a0, bu, u0);
    u1 = WMMA_BF16(a1, bu, u1);
  }
  int col = n0 + (lane & 15);
  int rb0 = m0 + ((lane >> 4) << 3);
#pragma unroll
  for (int r = 0; r < 8; ++r) {
    act[(size_t)(rb0 + r)      * N + col] = (bf16)(gelu_tanh(g0[r]) * u0[r]);
    act[(size_t)(rb0 + 16 + r) * N + col] = (bf16)(gelu_tanh(g1[r]) * u1[r]);
  }
}

// RoPE, faithful to the repo transpose quirk: angle position = batch index b
// (position_ids[s,b] == b since B==S). X is [4096][nheads*256]; in-place.
__global__ __launch_bounds__(256)
void rope_kernel(float* __restrict__ X, const float* __restrict__ inv_freq,
                 const int* __restrict__ pos_ids, int nheads) {
  int idx = blockIdx.x * blockDim.x + threadIdx.x;
  int total = NM * nheads * 128;
  if (idx >= total) return;
  int d   = idx & 127;
  int hh  = (idx >> 7) % nheads;
  int row = idx / (128 * nheads);
  int b = row >> 6, s = row & 63;
  float p   = (float)pos_ids[(s << 6) | b];   // == b (repo's 0<->2 transpose)
  float ang = p * inv_freq[d];
  float c = __cosf(ang), sn = __sinf(ang);
  size_t base = (size_t)row * (nheads * NHD) + hh * NHD + d;
  float x1 = X[base], x2 = X[base + 128];
  X[base]       = x1 * c - x2 * sn;
  X[base + 128] = x2 * c + x1 * sn;
}

// Fused attention, one workgroup per (b,h): scores via WMMA, fp16-style softmax,
// P@V via WMMA. Writes attn_out (B,S,H*HD) and probs to d_out's second segment.
__global__ __launch_bounds__(256)
void attention_kernel(const float* __restrict__ Q, const float* __restrict__ Kb,
                      const float* __restrict__ Vb, const float* __restrict__ mask,
                      float* __restrict__ attn_out, float* __restrict__ attn_probs) {
  __shared__ float sc[NS * NS];
  int bh = blockIdx.x, b = bh >> 3, h = bh & 7;
  int lane = threadIdx.x & 31, wave = threadIdx.x >> 5;
  const float* q   = Q  + (size_t)b * NS * ND + h * NHD;   // lda = 2048
  const float* k   = Kb + (size_t)b * NS * NHD;            // [64][256]
  const float* v   = Vb + (size_t)b * NS * NHD;
  const float* msk = mask + (size_t)b * NS * NS;

  // scores = q k^T / 16 + mask   (16 tiles of 16x16, K=256)
  for (int t = wave; t < 16; t += 8) {
    int mt = (t >> 2) * 16, nt = (t & 3) * 16;
    f32x8 acc = {};
    for (int kk = 0; kk < NHD; kk += 32) {
      bf16x16 a  = load_a(q, ND, mt, kk, lane);
      bf16x16 bb = load_b_wt(k, NHD, nt, kk, lane);
      acc = WMMA_BF16(a, bb, acc);
    }
    int col = nt + (lane & 15), rbase = mt + ((lane >> 4) << 3);
#pragma unroll
    for (int r = 0; r < 8; ++r) {
      int rr = rbase + r;
      sc[rr * NS + col] = acc[r] * 0.0625f + msk[rr * NS + col];
    }
  }
  __syncthreads();

  // softmax with float16 semantics (reference casts to fp16, softmaxes, casts back)
  if (threadIdx.x < NS) {
    float* row = sc + threadIdx.x * NS;
    float mx = -65504.f;
#pragma unroll 4
    for (int j = 0; j < NS; ++j) { float xv = (float)(_Float16)row[j]; row[j] = xv; mx = fmaxf(mx, xv); }
    float sum = 0.f;
#pragma unroll 4
    for (int j = 0; j < NS; ++j) {
      float e = (float)(_Float16)__expf((float)(_Float16)(row[j] - mx));
      row[j] = e; sum += e;
    }
    float inv = 1.0f / (float)(_Float16)sum;
#pragma unroll 4
    for (int j = 0; j < NS; ++j) {
      float pv = (float)(_Float16)(row[j] * inv);
      row[j] = pv;
      attn_probs[((size_t)bh * NS + threadIdx.x) * NS + j] = pv;
    }
  }
  __syncthreads();

  // out = P(64x64) @ V(64x256)   (64 tiles, 8 per wave, K=64)
  for (int t = wave; t < 64; t += 8) {
    int mt = (t >> 4) * 16, nt = (t & 15) * 16;
    f32x8 acc = {};
    for (int kk = 0; kk < NS; kk += 32) {
      bf16x16 a  = load_a((const float*)sc, NS, mt, kk, lane);
      bf16x16 bb = load_b_kn(v, NHD, nt, kk, lane);
      acc = WMMA_BF16(a, bb, acc);
    }
    int col = nt + (lane & 15), rbase = mt + ((lane >> 4) << 3);
#pragma unroll
    for (int r = 0; r < 8; ++r)
      attn_out[(size_t)(b * NS + rbase + r) * ND + h * NHD + col] = acc[r];
  }
}

// ---------------- host launch ----------------

extern "C" void kernel_launch(void* const* d_in, const int* in_sizes, int n_in,
                              void* d_out, int out_size, void* d_ws, size_t ws_size,
                              hipStream_t stream) {
  const float* hidden = (const float*)d_in[0];
  const float* amask  = (const float*)d_in[1];
  const int*   posids = (const int*)d_in[2];
  const float* invf   = (const float*)d_in[3];
  const float* wq     = (const float*)d_in[4];
  const float* wk     = (const float*)d_in[5];
  const float* wv     = (const float*)d_in[6];
  const float* wo     = (const float*)d_in[7];
  const float* wgate  = (const float*)d_in[8];
  const float* wup    = (const float*)d_in[9];
  const float* wdown  = (const float*)d_in[10];
  const float* ln1w   = (const float*)d_in[11];
  const float* ln2w   = (const float*)d_in[12];

  float* out_hidden = (float*)d_out;                       // B*S*D
  float* out_attn   = out_hidden + (size_t)NM * ND;        // B*H*S*S

  char* p = (char*)d_ws;
  float* h_norm = (float*)p; p += (size_t)NM * ND  * 4;    // 32 MB
  float* qb     = (float*)p; p += (size_t)NM * ND  * 4;    // 32 MB
  float* kb     = (float*)p; p += (size_t)NM * NHD * 4;    //  4 MB
  float* vb     = (float*)p; p += (size_t)NM * NHD * 4;    //  4 MB
  float* ao     = (float*)p; p += (size_t)NM * ND  * 4;    // 32 MB
  float* h2     = (float*)p; p += (size_t)NM * ND  * 4;    // 32 MB
  float* h2n    = (float*)p; p += (size_t)NM * ND  * 4;    // 32 MB
  bf16*  act    = (bf16*)p;  p += (size_t)NM * NI  * 2;    // 128 MB

  // 1. rmsnorm
  rmsnorm_kernel<<<NM, 256, 0, stream>>>(hidden, ln1w, h_norm, ND);
  // 2. q/k/v projections (grid.x = M-tiles, fastest, for L2 weight-stripe reuse)
  wmma_gemm<float><<<dim3(NM / 32, ND / 256),  256, 0, stream>>>(h_norm, wq, nullptr, qb, NM, ND,  ND);
  wmma_gemm<float><<<dim3(NM / 32, NHD / 256), 256, 0, stream>>>(h_norm, wk, nullptr, kb, NM, NHD, ND);
  wmma_gemm<float><<<dim3(NM / 32, NHD / 256), 256, 0, stream>>>(h_norm, wv, nullptr, vb, NM, NHD, ND);
  // 3. RoPE (in-place on scratch q/k)
  rope_kernel<<<(NM * NH * 128 + 255) / 256, 256, 0, stream>>>(qb, invf, posids, NH);
  rope_kernel<<<(NM * 1  * 128 + 255) / 256, 256, 0, stream>>>(kb, invf, posids, 1);
  // 4. attention (also writes probs into d_out's second segment)
  attention_kernel<<<NB * NH, 256, 0, stream>>>(qb, kb, vb, amask, ao, out_attn);
  // 5. o-proj + residual
  wmma_gemm<float><<<dim3(NM / 32, ND / 256), 256, 0, stream>>>(ao, wo, hidden, h2, NM, ND, ND);
  // 6. rmsnorm 2
  rmsnorm_kernel<<<NM, 256, 0, stream>>>(h2, ln2w, h2n, ND);
  // 7. fused gate/up + gelu-mul -> bf16 activation
  wmma_gateup<<<dim3(NM / 32, NI / 128), 256, 0, stream>>>(h2n, wgate, wup, act, NM, NI, ND);
  // 8. down-proj + residual -> final hidden output
  wmma_gemm<bf16><<<dim3(NM / 32, ND / 256), 256, 0, stream>>>(act, wdown, h2, out_hidden, NM, ND, NI);
}
